// DeepseekV2Layer_15882789061251
// MI455X (gfx1250) — compile-verified, matrix operand order checked by reading, not run on hardware
//
#include <hip/hip_runtime.h>
#include <cstdint>
#include <cstddef>

// ---------------------------------------------------------------------------
// DeepSeek-V2 layer on gfx1250 (MI455X): every matmul goes through
// v_wmma_f32_16x16x32_bf16 (f32 -> bf16 conversion while staging into LDS,
// f32 accumulation). 128x64 block tile, 256 threads = 8 wave32, each wave
// computes 32x32 via 2x2 WMMA. Double-buffered LDS pipeline (one barrier per
// K-step), 32-bit intra-tensor addressing, and a through-LDS vectorized
// epilogue (b128 coalesced stores). All shapes satisfy M%128==0, N%64==0,
// K%32==0 so staging is unguarded and fully vectorized.
// ---------------------------------------------------------------------------

typedef __bf16 bf16_t;
typedef __attribute__((ext_vector_type(16))) __bf16 v16bf;
typedef __attribute__((ext_vector_type(8)))  __bf16 v8bf;
typedef __attribute__((ext_vector_type(4)))  __bf16 v4bf;
typedef __attribute__((ext_vector_type(8)))  float  v8f;
typedef __attribute__((ext_vector_type(4)))  float  v4f;

namespace cfg {
constexpr int T  = 1024;
constexpr int H  = 2048;
constexpr int NH = 16;
constexpr int DN = 128;
constexpr int DR = 64;
constexpr int DV = 128;
constexpr int R  = 512;
constexpr int E  = 8;
constexpr int F  = 1408;
constexpr int FS = 2816;
constexpr float EPS   = 1e-6f;
constexpr float SCALE = 0.07216878364870323f;  // (DN+DR)^-0.5 = 192^-0.5
constexpr float LN_ROPE_BASE = 9.210340371976184f; // ln(10000)
}

// ---------------------------------------------------------------------------
// Generic tiled GEMM:  C[M,N] = A[M,K] * B            (f32 in, bf16 WMMA, f32 out)
//   BT == 0 : B stored K x N row-major;  BT == 1 : B stored N x K (C = A*B^T)
//   EPI == 0: C = AB
//   EPI == 1: C = AB + Res[m*ldres + n]
//   EPI == 2: atomicAdd(C, Scale[m*scaleStride + batch] * AB)
// ---------------------------------------------------------------------------
struct GemmP {
  const float* A; const float* B; float* C;
  const float* Res; const float* Scale;
  int M, N, K;
  int lda, ldb, ldc, ldres, scaleStride;
  long long sA, sB, sC;   // per-batch element strides (blockIdx.z)
};

template<int BT, int EPI>
__global__ __launch_bounds__(256) void gemm_bf16_wmma(GemmP p) {
  // 32 KB: two ping-pong staging buffers (12 KB each), whole block reused as
  // the 128x64 f32 output tile for the vectorized epilogue.
  __shared__ __align__(16) char smem[32768];

  const int tid   = threadIdx.x;
  const int lane  = tid & 31;
  const int wid   = tid >> 5;            // 0..7
  const int wm    = wid >> 1;            // 0..3 : wave tile row (32 rows each)
  const int wn    = wid & 1;             // 0..1 : wave tile col (32 cols each)
  const int l15   = lane & 15;
  const int kHalf = (lane >> 4) << 3;    // 0 or 8 (K offset for hi lanes)
  const int mHi   = kHalf;               // 0 or 8 (M offset for C/D hi lanes)

  const int mBase = blockIdx.y * 128;
  const int nBase = blockIdx.x * 64;
  const int z     = blockIdx.z;

  const float* Ab = p.A + (size_t)z * (size_t)p.sA;   // 64-bit once per block
  const float* Bb = p.B + (size_t)z * (size_t)p.sB;
  float*       Cb = p.C + (size_t)z * (size_t)p.sC;

  // ---- staging coordinates (all 32-bit) ----
  const int rowA = tid >> 3;             // 0..31 (i adds 32)
  const int colA = (tid & 7) * 4;
  const float* aSrc = Ab + ((mBase + rowA) * p.lda + colA);
  const int kB = tid >> 4;               // NN: 0..15 (i adds 16)
  const int nB = (tid & 15) * 4;
  const float* bSrcNN = Bb + (kB * p.ldb + nBase + nB);
  const int nT = tid >> 3;               // NT: 0..31 (i adds 32)
  const int cT = (tid & 7) * 4;
  const float* bSrcNT = Bb + ((nBase + nT) * p.ldb + cT);

  v8f acc[2][2] = {};
  v4f av[4];
  v4f bv[2];

  auto loadG = [&](int s) {
    const int kk = s * 32;
#pragma unroll
    for (int i = 0; i < 4; ++i)
      av[i] = *(const v4f*)(aSrc + (kk + i * 32 * p.lda));
    if (BT) {
#pragma unroll
      for (int i = 0; i < 2; ++i)
        bv[i] = *(const v4f*)(bSrcNT + (kk + i * 32 * p.ldb));
    } else {
#pragma unroll
      for (int i = 0; i < 2; ++i)
        bv[i] = *(const v4f*)(bSrcNN + (kk + i * 16) * p.ldb);
    }
    // speculative prefetch two tiles ahead (OOB silently dropped)
    __builtin_prefetch(aSrc + (kk + 64), 0, 3);
  };

  auto storeL = [&](int buf) {
    bf16_t* As = (bf16_t*)(smem + buf * 12288);
    bf16_t* Bs = (bf16_t*)(smem + buf * 12288 + 8192);
#pragma unroll
    for (int i = 0; i < 4; ++i) {
      v4bf w = { (bf16_t)av[i][0], (bf16_t)av[i][1],
                 (bf16_t)av[i][2], (bf16_t)av[i][3] };
      *(v4bf*)&As[(rowA + i * 32) * 32 + colA] = w;
    }
    if (BT) {
#pragma unroll
      for (int i = 0; i < 2; ++i) {
        v4bf w = { (bf16_t)bv[i][0], (bf16_t)bv[i][1],
                   (bf16_t)bv[i][2], (bf16_t)bv[i][3] };
        *(v4bf*)&Bs[(nT + i * 32) * 32 + cT] = w;
      }
    } else {   // transpose K x N -> [n][k]
#pragma unroll
      for (int i = 0; i < 2; ++i)
#pragma unroll
        for (int j = 0; j < 4; ++j)
          Bs[(nB + j) * 32 + kB + i * 16] = (bf16_t)bv[i][j];
    }
  };

  auto comp = [&](int buf) {
    const bf16_t* As = (const bf16_t*)(smem + buf * 12288);
    const bf16_t* Bs = (const bf16_t*)(smem + buf * 12288 + 8192);
    union FU { v16bf f; v8bf h[2]; };
    v16bf afrag[2], bfrag[2];
#pragma unroll
    for (int mt = 0; mt < 2; ++mt) {
      const bf16_t* ptr = &As[(wm * 32 + mt * 16 + l15) * 32 + kHalf];
      FU u; u.h[0] = *(const v8bf*)ptr; u.h[1] = *(const v8bf*)(ptr + 16);
      afrag[mt] = u.f;
    }
#pragma unroll
    for (int nt = 0; nt < 2; ++nt) {
      const bf16_t* ptr = &Bs[(wn * 32 + nt * 16 + l15) * 32 + kHalf];
      FU u; u.h[0] = *(const v8bf*)ptr; u.h[1] = *(const v8bf*)(ptr + 16);
      bfrag[nt] = u.f;
    }
#pragma unroll
    for (int mt = 0; mt < 2; ++mt)
#pragma unroll
      for (int nt = 0; nt < 2; ++nt)
        acc[mt][nt] = __builtin_amdgcn_wmma_f32_16x16x32_bf16(
            false, afrag[mt], false, bfrag[nt], (short)0, acc[mt][nt],
            false, false);
  };

  // ---- double-buffered pipeline: one barrier per K-step ----
  const int S = p.K >> 5;
  loadG(0);
  storeL(0);
  for (int s = 0; s < S; ++s) {
    __syncthreads();
    const bool more = (s + 1 < S);
    if (more) loadG(s + 1);
    comp(s & 1);
    if (more) storeL((s + 1) & 1);
  }
  __syncthreads();   // all fragment reads done before Cs overwrites staging

  // ---- epilogue through LDS: coalesced b128 stores ----
  float* Cs = (float*)smem;   // 128 x 64 f32 tile
#pragma unroll
  for (int mt = 0; mt < 2; ++mt) {
#pragma unroll
    for (int nt = 0; nt < 2; ++nt) {
      const int n = wn * 32 + nt * 16 + l15;
      const int mt0 = wm * 32 + mt * 16 + mHi;
#pragma unroll
      for (int r = 0; r < 8; ++r)
        Cs[(mt0 + r) * 64 + n] = acc[mt][nt][r];
    }
  }
  __syncthreads();

  const int mOut = tid >> 4;             // 0..15 (i adds 16)
  const int cOut = (tid & 15) * 4;
#pragma unroll
  for (int i = 0; i < 8; ++i) {
    const int m  = mOut + i * 16;
    const int gm = mBase + m;
    const int gn = nBase + cOut;
    v4f v = *(const v4f*)&Cs[m * 64 + cOut];
    if (EPI == 1) {
      v4f rv = *(const v4f*)(p.Res + (gm * p.ldres + gn));
      v += rv;
      *(v4f*)(Cb + (gm * p.ldc + gn)) = v;
    } else if (EPI == 2) {
      const float s = p.Scale[gm * p.scaleStride + z];
      float* cp = Cb + (gm * p.ldc + gn);
      atomicAdd(cp + 0, s * v[0]);
      atomicAdd(cp + 1, s * v[1]);
      atomicAdd(cp + 2, s * v[2]);
      atomicAdd(cp + 3, s * v[3]);
    } else {
      *(v4f*)(Cb + (gm * p.ldc + gn)) = v;
    }
  }
}

// ---------------------------------------------------------------------------
// RMSNorm: one block per row
// ---------------------------------------------------------------------------
__global__ __launch_bounds__(256) void rmsnorm_kernel(
    const float* __restrict__ in, const float* __restrict__ w,
    float* __restrict__ out, int len, int inStride, int outStride) {
  const int row = blockIdx.x, tid = threadIdx.x;
  const float* ir = in + (size_t)row * inStride;
  float ss = 0.f;
  for (int i = tid; i < len; i += 256) { float v = ir[i]; ss += v * v; }
  __shared__ float red[256];
  red[tid] = ss; __syncthreads();
  for (int off = 128; off; off >>= 1) {
    if (tid < off) red[tid] += red[tid + off];
    __syncthreads();
  }
  const float inv = rsqrtf(red[0] / (float)len + cfg::EPS);
  for (int i = tid; i < len; i += 256)
    out[(size_t)row * outStride + i] = ir[i] * inv * w[i];
}

// ---------------------------------------------------------------------------
// RoPE in place: rows of [ ..., pe(64) ] with half-rotation (32 | 32)
// grid.x = T, block = nHeads*32
// ---------------------------------------------------------------------------
__global__ void rope_kernel(float* buf, int rowStride, int headStride,
                            int peOffset) {
  const int t  = blockIdx.x;
  const int hh = threadIdx.x >> 5;
  const int i  = threadIdx.x & 31;
  const float inv = __expf(-(float)i * (cfg::LN_ROPE_BASE / 32.f));
  const float ang = (float)t * inv;
  const float c = cosf(ang), s = sinf(ang);
  float* p = buf + (size_t)t * rowStride + (size_t)hh * headStride + peOffset;
  const float x1 = p[i], x2 = p[i + 32];
  p[i]      = x1 * c - x2 * s;
  p[i + 32] = x2 * c + x1 * s;
}

// ---------------------------------------------------------------------------
// Pack k = concat(k_nope (from kv), roped k_pe broadcast (from kva)):
// k[t,h,0:128] = kv[t, h*256 + d]; k[t,h,128+j] = kva[t*576 + 512 + j]
// ---------------------------------------------------------------------------
__global__ void pack_k_kernel(const float* __restrict__ kv,
                              const float* __restrict__ kva,
                              float* __restrict__ k) {
  const long long total = (long long)cfg::T * cfg::NH * 192;
  long long idx = (long long)blockIdx.x * 256 + threadIdx.x;
  if (idx >= total) return;
  int d = (int)(idx % 192);
  int h = (int)((idx / 192) % cfg::NH);
  long long t = idx / (192 * cfg::NH);
  float v;
  if (d < 128) v = kv[t * 4096 + h * 256 + d];
  else         v = kva[t * 576 + 512 + (d - 128)];
  k[idx] = v;
}

// ---------------------------------------------------------------------------
// Causal softmax over scores[h][t][0:T], scale applied before max.
// grid = (T, NH), block = 256, T = 1024 fixed.
// ---------------------------------------------------------------------------
__global__ __launch_bounds__(256) void softmax_causal_kernel(float* scores) {
  const int t = blockIdx.x, h = blockIdx.y, tid = threadIdx.x;
  float* row = scores + ((size_t)h * cfg::T + t) * cfg::T;
  __shared__ float red[256];
  float vals[4];
  float vmax = -1e30f;
#pragma unroll
  for (int i = 0; i < 4; ++i) {
    int s = tid + i * 256;
    float v = (s <= t) ? row[s] * cfg::SCALE : -1e30f;
    vals[i] = v;
    vmax = fmaxf(vmax, v);
  }
  red[tid] = vmax; __syncthreads();
  for (int off = 128; off; off >>= 1) {
    if (tid < off) red[tid] = fmaxf(red[tid], red[tid + off]);
    __syncthreads();
  }
  const float m = red[0]; __syncthreads();
  float sum = 0.f;
#pragma unroll
  for (int i = 0; i < 4; ++i) {
    int s = tid + i * 256;
    float e = (s <= t) ? expf(vals[i] - m) : 0.f;
    vals[i] = e; sum += e;
  }
  red[tid] = sum; __syncthreads();
  for (int off = 128; off; off >>= 1) {
    if (tid < off) red[tid] += red[tid + off];
    __syncthreads();
  }
  const float inv = 1.f / red[0];
#pragma unroll
  for (int i = 0; i < 4; ++i) row[tid + i * 256] = vals[i] * inv;
}

// ---------------------------------------------------------------------------
// Gate logits: glog[t,e] = sum_k h2[t,k] * gate_w[k,e]; one wave per expert.
// grid = T, block = 256 (8 waves)
// ---------------------------------------------------------------------------
__global__ __launch_bounds__(256) void gate_logits_kernel(
    const float* __restrict__ h2, const float* __restrict__ gw,
    float* __restrict__ glog) {
  const int t = blockIdx.x;
  const int e = threadIdx.x >> 5;
  const int lane = threadIdx.x & 31;
  float s = 0.f;
  for (int k = lane; k < cfg::H; k += 32)
    s += h2[(size_t)t * cfg::H + k] * gw[(size_t)k * cfg::E + e];
  for (int off = 16; off; off >>= 1) s += __shfl_xor(s, off, 32);
  if (lane == 0) glog[(size_t)t * cfg::E + e] = s;
}

// ---------------------------------------------------------------------------
// Per-token gating: softmax(8) -> group max (G=4 of 2) -> top-2 groups ->
// top-2 masked experts -> combine weights (T x E)
// ---------------------------------------------------------------------------
__global__ void gating_kernel(const float* __restrict__ glog,
                              float* __restrict__ combine) {
  int t = blockIdx.x * blockDim.x + threadIdx.x;
  if (t >= cfg::T) return;
  float sc[8];
  float mx = -1e30f;
#pragma unroll
  for (int e = 0; e < 8; ++e) { sc[e] = glog[t * 8 + e]; mx = fmaxf(mx, sc[e]); }
  float sum = 0.f;
#pragma unroll
  for (int e = 0; e < 8; ++e) { sc[e] = expf(sc[e] - mx); sum += sc[e]; }
#pragma unroll
  for (int e = 0; e < 8; ++e) sc[e] /= sum;
  float g[4];
#pragma unroll
  for (int gi = 0; gi < 4; ++gi) g[gi] = fmaxf(sc[2 * gi], sc[2 * gi + 1]);
  int g1 = 0;
#pragma unroll
  for (int gi = 1; gi < 4; ++gi) if (g[gi] > g[g1]) g1 = gi;
  int g2 = -1;
#pragma unroll
  for (int gi = 0; gi < 4; ++gi)
    if (gi != g1 && (g2 < 0 || g[gi] > g[g2])) g2 = gi;
  float msk[8];
#pragma unroll
  for (int e = 0; e < 8; ++e) {
    int grp = e >> 1;
    msk[e] = (grp == g1 || grp == g2) ? sc[e] : 0.f;
  }
  int i1 = 0;
#pragma unroll
  for (int e = 1; e < 8; ++e) if (msk[e] > msk[i1]) i1 = e;
  int i2 = -1;
#pragma unroll
  for (int e = 0; e < 8; ++e)
    if (e != i1 && (i2 < 0 || msk[e] > msk[i2])) i2 = e;
#pragma unroll
  for (int e = 0; e < 8; ++e)
    combine[t * 8 + e] = (e == i1) ? msk[i1] : (e == i2) ? msk[i2] : 0.f;
}

// ---------------------------------------------------------------------------
// out[r,f] = silu(in[r,f]) * in[r,F+f]   (row length of `in` is 2F)
// ---------------------------------------------------------------------------
__global__ void silu_mul_kernel(const float* __restrict__ in,
                                float* __restrict__ out,
                                long long nRows, int Fdim) {
  long long idx = (long long)blockIdx.x * 256 + threadIdx.x;
  long long total = nRows * (long long)Fdim;
  if (idx >= total) return;
  long long r = idx / Fdim;
  int f = (int)(idx % Fdim);
  const float* row = in + r * (2LL * Fdim);
  float g = row[f], u = row[Fdim + f];
  out[r * (long long)Fdim + f] = g * u / (1.f + expf(-g));
}

// ---------------------------------------------------------------------------
// out = a + b (final: shout(+x1 already fused) + routed)
// ---------------------------------------------------------------------------
__global__ void add2_kernel(float* __restrict__ out,
                            const float* __restrict__ a,
                            const float* __restrict__ b, long long n) {
  long long i = (long long)blockIdx.x * 256 + threadIdx.x;
  if (i < n) out[i] = a[i] + b[i];
}

// ---------------------------------------------------------------------------
// Host side
// ---------------------------------------------------------------------------
static inline GemmP mkp(const float* A, const float* B, float* C,
                        int M, int N, int K, int lda, int ldb, int ldc,
                        long long sA = 0, long long sB = 0, long long sC = 0,
                        const float* Res = nullptr, int ldres = 0,
                        const float* Scale = nullptr, int scaleStride = 0) {
  GemmP p;
  p.A = A; p.B = B; p.C = C; p.Res = Res; p.Scale = Scale;
  p.M = M; p.N = N; p.K = K;
  p.lda = lda; p.ldb = ldb; p.ldc = ldc;
  p.ldres = ldres; p.scaleStride = scaleStride;
  p.sA = sA; p.sB = sB; p.sC = sC;
  return p;
}
static inline dim3 ggrid(int M, int N, int batch) {
  return dim3((unsigned)(N / 64), (unsigned)(M / 128), (unsigned)batch);
}

extern "C" void kernel_launch(void* const* d_in, const int* in_sizes, int n_in,
                              void* d_out, int out_size, void* d_ws,
                              size_t ws_size, hipStream_t stream) {
  using namespace cfg;
  (void)in_sizes; (void)n_in; (void)out_size; (void)ws_size;

  const float* x          = (const float*)d_in[0];
  const float* norm1_w    = (const float*)d_in[1];
  const float* norm2_w    = (const float*)d_in[2];
  const float* q_w        = (const float*)d_in[3];   // (H, NH*192)
  const float* kv_a_w     = (const float*)d_in[4];   // (H, R+DR)
  const float* kv_a_nw    = (const float*)d_in[5];   // (R,)
  const float* kv_b_w     = (const float*)d_in[6];   // (R, NH*256)
  const float* o_w        = (const float*)d_in[7];   // (NH*DV, H)
  const float* gate_w     = (const float*)d_in[8];   // (H, E)
  const float* exp_gu_w   = (const float*)d_in[9];   // (E, H, 2F)
  const float* exp_dn_w   = (const float*)d_in[10];  // (E, F, H)
  const float* sh_gu_w    = (const float*)d_in[11];  // (H, 2FS)
  const float* sh_dn_w    = (const float*)d_in[12];  // (FS, H)
  float* out = (float*)d_out;

  // workspace carve-out (f32, 256B aligned)
  char* ws = (char*)d_ws;
  size_t off = 0;
  auto alloc = [&](size_t elems) -> float* {
    float* p = (float*)(ws + off);
    off += ((elems * sizeof(float) + 255) / 256) * 256;
    return p;
  };
  float* h_buf   = alloc((size_t)T * H);            // rmsnorm1(x)
  float* q_buf   = alloc((size_t)T * NH * 192);     // q (= qf after in-place rope)
  float* kva     = alloc((size_t)T * (R + DR));     // kv_a output
  float* kvcn    = alloc((size_t)T * R);            // rmsnorm(kv_c)
  float* kv_buf  = alloc((size_t)T * NH * 256);     // kv_b output (k_nope | v)
  float* k_buf   = alloc((size_t)T * NH * 192);     // packed k
  float* scores  = alloc((size_t)NH * T * T);       // logits -> p (in place)
  float* attn    = alloc((size_t)T * NH * DV);
  float* x1      = alloc((size_t)T * H);
  float* h2      = alloc((size_t)T * H);
  float* glog    = alloc((size_t)T * E);
  float* combine = alloc((size_t)T * E);
  float* gu      = alloc((size_t)E * T * 2 * F);
  float* act     = alloc((size_t)E * T * F);
  float* routed  = alloc((size_t)T * H);
  float* sgu     = alloc((size_t)T * 2 * FS);
  float* shact   = alloc((size_t)T * FS);
  float* shout   = alloc((size_t)T * H);

  // 1. h = rmsnorm(x) * norm1_w
  rmsnorm_kernel<<<T, 256, 0, stream>>>(x, norm1_w, h_buf, H, H, H);

  // 2. q = h @ q_w   (1024 x 3072, K=2048)
  gemm_bf16_wmma<0, 0><<<ggrid(T, NH * 192, 1), 256, 0, stream>>>(
      mkp(h_buf, q_w, q_buf, T, NH * 192, H, H, NH * 192, NH * 192));

  // 3. kva = h @ kv_a_w  (1024 x 576)
  gemm_bf16_wmma<0, 0><<<ggrid(T, R + DR, 1), 256, 0, stream>>>(
      mkp(h_buf, kv_a_w, kva, T, R + DR, H, H, R + DR, R + DR));

  // 4. kv_c = rmsnorm(kva[:, :512]) * kv_a_norm_w
  rmsnorm_kernel<<<T, 256, 0, stream>>>(kva, kv_a_nw, kvcn, R, R + DR, R);

  // 5. kv = kv_c @ kv_b_w  (1024 x 4096, K=512)
  gemm_bf16_wmma<0, 0><<<ggrid(T, NH * 256, 1), 256, 0, stream>>>(
      mkp(kvcn, kv_b_w, kv_buf, T, NH * 256, R, R, NH * 256, NH * 256));

  // 6/7. rope q_pe (in place inside q) and k_pe (in place inside kva)
  rope_kernel<<<T, NH * 32, 0, stream>>>(q_buf, NH * 192, 192, DN);
  rope_kernel<<<T, 32, 0, stream>>>(kva, R + DR, 0, R);

  // 8. pack k = [k_nope | k_pe broadcast]  (T, NH, 192)
  {
    long long total = (long long)T * NH * 192;
    pack_k_kernel<<<(unsigned)((total + 255) / 256), 256, 0, stream>>>(
        kv_buf, kva, k_buf);
  }

  // 9. scores[h] = qf[h] @ k[h]^T  (batched over heads, B transposed)
  gemm_bf16_wmma<1, 0><<<ggrid(T, T, NH), 256, 0, stream>>>(
      mkp(q_buf, k_buf, scores, T, T, 192, NH * 192, NH * 192, T,
          192, 192, (long long)T * T));

  // 10. causal softmax (scale inside)
  softmax_causal_kernel<<<dim3(T, NH), 256, 0, stream>>>(scores);

  // 11. attn[h] = p[h] @ v[h]   (v strided inside kv_buf at +128)
  gemm_bf16_wmma<0, 0><<<ggrid(T, DV, NH), 256, 0, stream>>>(
      mkp(scores, kv_buf + DN, attn, T, DV, T, T, NH * 256, NH * DV,
          (long long)T * T, 256, DV));

  // 12. x1 = x + attn @ o_w
  gemm_bf16_wmma<0, 1><<<ggrid(T, H, 1), 256, 0, stream>>>(
      mkp(attn, o_w, x1, T, H, NH * DV, NH * DV, H, H, 0, 0, 0, x, H));

  // 13. h2 = rmsnorm(x1) * norm2_w
  rmsnorm_kernel<<<T, 256, 0, stream>>>(x1, norm2_w, h2, H, H, H);

  // 14/15. gating
  gate_logits_kernel<<<T, 256, 0, stream>>>(h2, gate_w, glog);
  gating_kernel<<<(T + 255) / 256, 256, 0, stream>>>(glog, combine);

  // 16. gu[e] = h2 @ exp_gate_up[e]  (batch 8, A shared)
  gemm_bf16_wmma<0, 0><<<ggrid(T, 2 * F, E), 256, 0, stream>>>(
      mkp(h2, exp_gu_w, gu, T, 2 * F, H, H, 2 * F, 2 * F,
          0, (long long)H * 2 * F, (long long)T * 2 * F));

  // 17. act = silu(gu[:F]) * gu[F:]
  {
    long long total = (long long)E * T * F;
    silu_mul_kernel<<<(unsigned)((total + 255) / 256), 256, 0, stream>>>(
        gu, act, (long long)E * T, F);
  }

  // 18. routed = 0; 19. routed += combine[t,e] * (act[e] @ exp_down[e])
  hipMemsetAsync(routed, 0, (size_t)T * H * sizeof(float), stream);
  gemm_bf16_wmma<0, 2><<<ggrid(T, H, E), 256, 0, stream>>>(
      mkp(act, exp_dn_w, routed, T, H, F, F, H, H,
          (long long)T * F, (long long)F * H, 0, nullptr, 0, combine, E));

  // 20. sgu = h2 @ sh_gate_up  (1024 x 5632)
  gemm_bf16_wmma<0, 0><<<ggrid(T, 2 * FS, 1), 256, 0, stream>>>(
      mkp(h2, sh_gu_w, sgu, T, 2 * FS, H, H, 2 * FS, 2 * FS));

  // 21. shact = silu(sgu[:FS]) * sgu[FS:]
  {
    long long total = (long long)T * FS;
    silu_mul_kernel<<<(unsigned)((total + 255) / 256), 256, 0, stream>>>(
        sgu, shact, (long long)T, FS);
  }

  // 22. shout = shact @ sh_down + x1   (residual fused)
  gemm_bf16_wmma<0, 1><<<ggrid(T, H, 1), 256, 0, stream>>>(
      mkp(shact, sh_dn_w, shout, T, H, FS, FS, H, H, 0, 0, 0, x1, H));

  // 23. out = shout + routed  (= x1 + shared + routed)
  {
    long long total = (long long)T * H;
    add2_kernel<<<(unsigned)((total + 255) / 256), 256, 0, stream>>>(
        out, shout, routed, total);
  }
}
